// CompositionalAttention_83176336655076
// MI455X (gfx1250) — matrix operand order repaired and checked by reading, not compiled, not run on hardware
//
#include <hip/hip_runtime.h>
#include <hip/hip_bf16.h>

// ---------------- types ----------------
typedef __bf16 bf16;
typedef __attribute__((ext_vector_type(16))) __bf16       v16bf;
typedef __attribute__((ext_vector_type(8)))  float        v8f;
typedef __attribute__((ext_vector_type(4)))  unsigned int v4u;

#define DEV __device__ __forceinline__

// ---------------- problem constants ----------------
constexpr int BATCH  = 2;
constexpr int NSEQ   = 2048;
constexpr int DIMC   = 512;
constexpr int DH     = 64;
constexpr int SH     = 8;   // search heads
constexpr int RH     = 2;   // retrieval heads
constexpr int BN     = BATCH * NSEQ;          // 4096 rows of x
// ---------------- workspace layout (bytes) ----------------
constexpr size_t OFF_XBF   = 0;                                    // 4096*512 bf16
constexpr size_t OFF_WSQT  = OFF_XBF   + (size_t)BN*DIMC*2;        // 512*512
constexpr size_t OFF_WSKT  = OFF_WSQT  + (size_t)DIMC*DIMC*2;
constexpr size_t OFF_WRVT  = OFF_WSKT  + (size_t)DIMC*DIMC*2;      // 128*512
constexpr size_t OFF_WRQT  = OFF_WRVT  + (size_t)(RH*DH)*DIMC*2;
constexpr size_t OFF_WRKT  = OFF_WRQT  + (size_t)DIMC*DIMC*2;      // 64*64
constexpr size_t OFF_WOUTT = OFF_WRKT  + (size_t)DH*DH*2;
constexpr size_t OFF_SQ    = OFF_WOUTT + (size_t)DIMC*DIMC*2;      // (b,s,n,d) bf16
constexpr size_t OFF_SK    = OFF_SQ    + (size_t)BATCH*SH*NSEQ*DH*2;
constexpr size_t OFF_RK    = OFF_SQ;   // alias: RK reuses SQ+SK after flash
constexpr size_t OFF_RQ    = OFF_SK    + (size_t)BATCH*SH*NSEQ*DH*2;
constexpr size_t OFF_RVT   = OFF_RQ    + (size_t)BATCH*SH*NSEQ*DH*2; // (b,r,d,n)
constexpr size_t OFF_RET   = OFF_RVT   + (size_t)BATCH*RH*DH*NSEQ*2; // (b,s,r,n,d)
constexpr size_t OFF_O     = OFF_RET   + (size_t)BATCH*SH*RH*NSEQ*DH*2; // (b,n,s,d)

// ---------------- WMMA helpers ----------------
DEV v8f wmma_bf16(v16bf a, v16bf b, v8f c) {
  return __builtin_amdgcn_wmma_f32_16x16x32_bf16(false, a, false, b, (short)0, c,
                                                 false, false);
}

union FragBF { v16bf bf; v4u u[2]; };

// A fragment (16x32, M x K), row-major source, row stride ld (elements).
// lane l<16: row l, K = {0..7, 16..23}; lane l>=16: row l-16, K = {8..15, 24..31}
DEV v16bf load_a_frag(const bf16* base, int ld, int m0, int k0, int lane) {
  int half = lane >> 4, m = lane & 15;
  const bf16* p = base + (size_t)(m0 + m) * ld + k0 + 8 * half;
  FragBF f;
  f.u[0] = *(const v4u*)(p);
  f.u[1] = *(const v4u*)(p + 16);
  return f.bf;
}

// B fragment (32x16, K x N) from a column-major store BT[n][k], row stride ldbt.
// lane l<16: col l, K = {0..7, 16..23}; lane l>=16: col l-16, K = {8..15, 24..31}
DEV v16bf load_b_frag(const bf16* bt, int ldbt, int k0, int n0, int lane) {
  int khalf = lane >> 4, nn = lane & 15;
  const bf16* p = bt + (size_t)(n0 + nn) * ldbt + k0 + 8 * khalf;
  FragBF f;
  f.u[0] = *(const v4u*)(p);
  f.u[1] = *(const v4u*)(p + 16);
  return f.bf;
}

// C/D fragment store: lane holds col n0+(lane&15), rows m = m0 + v + 8*(lane>=16)
// mode 0: bf16 dst[row*N+col]           (plain, ldc = N)
// mode 1: bf16 heads: (b,h,i,d) with H heads, seq 2048
// mode 2: bf16 transposed rv: (b,r,d,i)
// mode 3: f32  dst[row*N+col]
DEV void store_c_frag(const v8f& c, int m0, int n0, int lane, int mode, int N,
                      int H, void* dst) {
  int col = n0 + (lane & 15);
  int half = lane >> 4;
#pragma unroll
  for (int v = 0; v < 8; ++v) {
    int row = m0 + v + 8 * half;
    float val = c[v];
    if (mode == 3) {
      ((float*)dst)[(size_t)row * N + col] = val;
    } else {
      size_t idx;
      if (mode == 0) {
        idx = (size_t)row * N + col;
      } else if (mode == 1) {
        int b_ = row >> 11, i_ = row & 2047, h_ = col >> 6, d_ = col & 63;
        idx = (((size_t)(b_ * H + h_) * NSEQ + i_) * DH + d_);
      } else {  // mode 2
        int b_ = row >> 11, i_ = row & 2047, r_ = col >> 6, d_ = col & 63;
        idx = (((size_t)(b_ * RH + r_) * DH + d_) * NSEQ + i_);
      }
      ((bf16*)dst)[idx] = (bf16)val;
    }
  }
}

// ---------------- pack kernels ----------------
__global__ void cast_bf16_kernel(const float* __restrict__ src,
                                 bf16* __restrict__ dst, int n) {
  int i = blockIdx.x * blockDim.x + threadIdx.x;
  if (i < n) dst[i] = (bf16)src[i];
}

// dst[c*K + k] = src[k*C + c] * scale  (transpose KxC -> CxK, fp32 -> bf16)
__global__ void transpose_scale_kernel(const float* __restrict__ src,
                                       bf16* __restrict__ dst, int K, int C,
                                       float scale) {
  int i = blockIdx.x * blockDim.x + threadIdx.x;
  if (i >= K * C) return;
  int c = i / K;
  int k = i - c * K;
  dst[i] = (bf16)(src[(size_t)k * C + c] * scale);
}

// ---------------- generic bf16 WMMA GEMM (ping-pong pipelined) -------------
// C[M,N] = A[M,K] @ B[K,N], B given transposed (BT[n][k]).
// 128 threads = 4 waves; block tile 64(M) x 64(N); wave tile 16 x 64.
// Two fragment sets, consumed and refilled in place (k-loop unrolled by 2):
//   wmma(set0) -> load set0(k+64) -> wmma(set1) -> load set1(k+96)
// Each WMMA group waits only for its own (older) loads; the other set's loads
// remain in flight. No cross-set register copies.
__global__ __launch_bounds__(128) void gemm_bf16_wmma(
    const bf16* __restrict__ A, int lda, const bf16* __restrict__ BT, int ldbt,
    int M, int N, int K, void* dst, int mode, int H) {
  int lane = threadIdx.x & 31;
  int w = threadIdx.x >> 5;
  int nb = N >> 6;
  int bm = blockIdx.x / nb;
  int bn = blockIdx.x - bm * nb;
  int m0 = bm * 64 + w * 16;
  int n0 = bn * 64;

  v8f acc[4] = {};
  v16bf a0 = load_a_frag(A, lda, m0, 0, lane);
  v16bf b0[4];
#pragma unroll
  for (int nt = 0; nt < 4; ++nt)
    b0[nt] = load_b_frag(BT, ldbt, 0, n0 + nt * 16, lane);
  v16bf a1 = load_a_frag(A, lda, m0, 32, lane);
  v16bf b1[4];
#pragma unroll
  for (int nt = 0; nt < 4; ++nt)
    b1[nt] = load_b_frag(BT, ldbt, 32, n0 + nt * 16, lane);

  for (int k0 = 0; k0 + 64 < K; k0 += 64) {
#pragma unroll
    for (int nt = 0; nt < 4; ++nt) acc[nt] = wmma_bf16(a0, b0[nt], acc[nt]);
    a0 = load_a_frag(A, lda, m0, k0 + 64, lane);
#pragma unroll
    for (int nt = 0; nt < 4; ++nt)
      b0[nt] = load_b_frag(BT, ldbt, k0 + 64, n0 + nt * 16, lane);
#pragma unroll
    for (int nt = 0; nt < 4; ++nt) acc[nt] = wmma_bf16(a1, b1[nt], acc[nt]);
    a1 = load_a_frag(A, lda, m0, k0 + 96, lane);
#pragma unroll
    for (int nt = 0; nt < 4; ++nt)
      b1[nt] = load_b_frag(BT, ldbt, k0 + 96, n0 + nt * 16, lane);
  }
#pragma unroll
  for (int nt = 0; nt < 4; ++nt) acc[nt] = wmma_bf16(a0, b0[nt], acc[nt]);
#pragma unroll
  for (int nt = 0; nt < 4; ++nt) acc[nt] = wmma_bf16(a1, b1[nt], acc[nt]);

#pragma unroll
  for (int nt = 0; nt < 4; ++nt)
    store_c_frag(acc[nt], m0, n0 + nt * 16, lane, mode, N, H, dst);
}

// ---------------- fused flash search attention ----------------
// Per (b,s) head: retrieved[r,i,d] = softmax_j(sq_i . sk_j) @ rv[r]
// Computes S^T = sk @ sq^T so softmax axis (j) lies along C-fragment rows:
// in-lane reduce over 8 values + one shfl_xor(16). Two-pass flash (m,l first,
// then normalized P), P staged via per-wave LDS to become an A operand.
DEV void flash_update_ml(const v8f& c, float& m, float& l) {
  float tmax = c[0];
#pragma unroll
  for (int v = 1; v < 8; ++v) tmax = fmaxf(tmax, c[v]);
  tmax = fmaxf(tmax, __shfl_xor(tmax, 16, 32));
  float nm = fmaxf(m, tmax);
  float ts = 0.f;
#pragma unroll
  for (int v = 0; v < 8; ++v) ts += __expf(c[v] - nm);
  ts += __shfl_xor(ts, 16, 32);
  l = l * __expf(m - nm) + ts;
  m = nm;
}

__global__ __launch_bounds__(256) void flash_search_kernel(
    const bf16* __restrict__ SQ, const bf16* __restrict__ SK,
    const bf16* __restrict__ RVT, bf16* __restrict__ RET) {
  __shared__ __align__(16) bf16 Pbuf[8][16 * 32];  // per-wave 16(i) x 32(j)

  int lane = threadIdx.x & 31;
  int w = threadIdx.x >> 5;
  int head = blockIdx.x >> 4;  // b*8+s
  int iblk = blockIdx.x & 15;
  int b = head >> 3;
  int i0 = iblk * 128 + w * 16;

  const bf16* sq = SQ + (size_t)head * NSEQ * DH;
  const bf16* sk = SK + (size_t)head * NSEQ * DH;
  const bf16* rvt = RVT + (size_t)b * RH * DH * NSEQ;

  // sq^T as B operand (column i = sq row i): constant across the j loop
  v16bf bq0 = load_b_frag(sq, DH, 0, i0, lane);
  v16bf bq1 = load_b_frag(sq, DH, 32, i0, lane);

  // ---- pass 1: running max + sum over j (ping-pong sk fragment sets) ----
  float m = -1e30f, l = 0.f;
  {
    v16bf a00 = load_a_frag(sk, DH, 0, 0, lane);
    v16bf a01 = load_a_frag(sk, DH, 0, 32, lane);
    v16bf a10 = load_a_frag(sk, DH, 16, 0, lane);
    v16bf a11 = load_a_frag(sk, DH, 16, 32, lane);
    for (int j0 = 0; j0 + 32 < NSEQ; j0 += 32) {
      v8f c = {};
      c = wmma_bf16(a00, bq0, c);
      c = wmma_bf16(a01, bq1, c);
      flash_update_ml(c, m, l);
      a00 = load_a_frag(sk, DH, j0 + 32, 0, lane);
      a01 = load_a_frag(sk, DH, j0 + 32, 32, lane);
      v8f c2 = {};
      c2 = wmma_bf16(a10, bq0, c2);
      c2 = wmma_bf16(a11, bq1, c2);
      flash_update_ml(c2, m, l);
      a10 = load_a_frag(sk, DH, j0 + 48, 0, lane);
      a11 = load_a_frag(sk, DH, j0 + 48, 32, lane);
    }
    v8f c = {};
    c = wmma_bf16(a00, bq0, c);
    c = wmma_bf16(a01, bq1, c);
    flash_update_ml(c, m, l);
    v8f c2 = {};
    c2 = wmma_bf16(a10, bq0, c2);
    c2 = wmma_bf16(a11, bq1, c2);
    flash_update_ml(c2, m, l);
  }
  float rinv = 1.0f / l;

  // ---- pass 2: P = exp(S - m)/l, retrieved[r] += P @ rv[r] ----
  v8f acc[RH][4] = {};
  bf16* pb = Pbuf[w];
  int half = lane >> 4, iloc = lane & 15;
  for (int j0 = 0; j0 < NSEQ; j0 += 32) {
    // sk A-frags first (consumed first), then rv B-frags (consumed after the
    // LDS round trip) -> S^T WMMAs only drain the sk loads.
    v16bf sa[2][2];
#pragma unroll
    for (int t = 0; t < 2; ++t) {
      sa[t][0] = load_a_frag(sk, DH, j0 + 16 * t, 0, lane);
      sa[t][1] = load_a_frag(sk, DH, j0 + 16 * t, 32, lane);
    }
    v16bf bv[RH][4];
#pragma unroll
    for (int r = 0; r < RH; ++r) {
      const bf16* rvh = rvt + (size_t)r * DH * NSEQ;
#pragma unroll
      for (int nt = 0; nt < 4; ++nt)
        bv[r][nt] = load_b_frag(rvh, NSEQ, j0, nt * 16, lane);
    }
#pragma unroll
    for (int t = 0; t < 2; ++t) {
      v8f c = {};
      c = wmma_bf16(sa[t][0], bq0, c);
      c = wmma_bf16(sa[t][1], bq1, c);
      // C layout: lane col = i (local), rows = j (local); transpose into LDS P[i][j]
#pragma unroll
      for (int v = 0; v < 8; ++v) {
        float p = __expf(c[v] - m) * rinv;
        pb[iloc * 32 + (16 * t + v + 8 * half)] = (bf16)p;
      }
    }
    // P tile (16 i x 32 j) as A operand from LDS (same-wave DS is in-order)
    v16bf pa = load_a_frag(pb, 32, 0, 0, lane);
#pragma unroll
    for (int r = 0; r < RH; ++r)
#pragma unroll
      for (int nt = 0; nt < 4; ++nt)
        acc[r][nt] = wmma_bf16(pa, bv[r][nt], acc[r][nt]);
  }

  // store retrieved (b,s,r,i,d) in bf16
#pragma unroll
  for (int r = 0; r < RH; ++r) {
    bf16* out = RET + (((size_t)head * RH + r) * NSEQ) * DH;
#pragma unroll
    for (int nt = 0; nt < 4; ++nt) {
      int d = nt * 16 + (lane & 15);
#pragma unroll
      for (int v = 0; v < 8; ++v) {
        int i = i0 + v + 8 * half;
        out[(size_t)i * DH + d] = (bf16)acc[r][nt][v];
      }
    }
  }
}

// ---------------- retrieval softmax over r=2 + combine ----------------
// 32 lanes per (b,s,i) row; lane handles d = lane and d = lane+32.
__global__ __launch_bounds__(256) void combine_kernel(
    const bf16* __restrict__ RQ, const bf16* __restrict__ RET,
    const bf16* __restrict__ RK, bf16* __restrict__ O) {
  int lane = threadIdx.x & 31;
  int row = blockIdx.x * 8 + (threadIdx.x >> 5);  // over b*s*n = 32768
  int hs = row >> 11;       // b*8+s
  int i = row & 2047;
  int b_ = row >> 14;
  int s_ = hs & 7;

  const bf16* rq = RQ + (size_t)row * DH;
  const bf16* rt0 = RET + (((size_t)hs * RH + 0) * NSEQ + i) * DH;
  const bf16* rt1 = RET + (((size_t)hs * RH + 1) * NSEQ + i) * DH;
  const bf16* rk0 = RK + (((size_t)hs * RH + 0) * NSEQ + i) * DH;
  const bf16* rk1 = RK + (((size_t)hs * RH + 1) * NSEQ + i) * DH;

  float q0 = (float)rq[lane], q1 = (float)rq[lane + 32];
  float p0 = q0 * (float)rk0[lane] + q1 * (float)rk0[lane + 32];
  float p1 = q0 * (float)rk1[lane] + q1 * (float)rk1[lane + 32];
#pragma unroll
  for (int ofs = 16; ofs > 0; ofs >>= 1) {
    p0 += __shfl_xor(p0, ofs, 32);
    p1 += __shfl_xor(p1, ofs, 32);
  }
  float a = fmaxf(p0, p1);
  float e0 = __expf(p0 - a), e1 = __expf(p1 - a);
  float w0 = e0 / (e0 + e1), w1 = 1.0f - w0;

  // O layout (b, i, s, d) so the final GEMM reads contiguous K = s*d
  size_t ob = (((size_t)b_ * NSEQ + i) * SH + s_) * DH;
  O[ob + lane] = (bf16)(w0 * (float)rt0[lane] + w1 * (float)rt1[lane]);
  O[ob + lane + 32] =
      (bf16)(w0 * (float)rt0[lane + 32] + w1 * (float)rt1[lane + 32]);
}

// ---------------- launcher ----------------
extern "C" void kernel_launch(void* const* d_in, const int* in_sizes, int n_in,
                              void* d_out, int out_size, void* d_ws,
                              size_t ws_size, hipStream_t stream) {
  (void)in_sizes; (void)n_in; (void)out_size; (void)ws_size;
  const float* x    = (const float*)d_in[0];
  const float* Wsq  = (const float*)d_in[1];
  const float* Wsk  = (const float*)d_in[2];
  const float* Wrv  = (const float*)d_in[3];
  const float* Wrq  = (const float*)d_in[4];
  const float* Wrk  = (const float*)d_in[5];
  const float* Wout = (const float*)d_in[6];

  char* ws = (char*)d_ws;
  bf16* XBF   = (bf16*)(ws + OFF_XBF);
  bf16* WSQT  = (bf16*)(ws + OFF_WSQT);
  bf16* WSKT  = (bf16*)(ws + OFF_WSKT);
  bf16* WRVT  = (bf16*)(ws + OFF_WRVT);
  bf16* WRQT  = (bf16*)(ws + OFF_WRQT);
  bf16* WRKT  = (bf16*)(ws + OFF_WRKT);
  bf16* WOUTT = (bf16*)(ws + OFF_WOUTT);
  bf16* SQ    = (bf16*)(ws + OFF_SQ);
  bf16* SK    = (bf16*)(ws + OFF_SK);
  bf16* RQ    = (bf16*)(ws + OFF_RQ);
  bf16* RVT   = (bf16*)(ws + OFF_RVT);
  bf16* RET   = (bf16*)(ws + OFF_RET);
  bf16* RK    = (bf16*)(ws + OFF_RK);   // aliases SQ+SK (dead after flash)
  bf16* O     = (bf16*)(ws + OFF_O);

  const float scale = 0.125f;  // 64^-0.5, folded into Wsq / Wrq

  // 1) pack: x -> bf16; weights -> transposed bf16 (B operands)
  cast_bf16_kernel<<<(BN * DIMC + 255) / 256, 256, 0, stream>>>(x, XBF,
                                                                BN * DIMC);
  transpose_scale_kernel<<<(DIMC * DIMC + 255) / 256, 256, 0, stream>>>(
      Wsq, WSQT, DIMC, DIMC, scale);
  transpose_scale_kernel<<<(DIMC * DIMC + 255) / 256, 256, 0, stream>>>(
      Wsk, WSKT, DIMC, DIMC, 1.0f);
  transpose_scale_kernel<<<(DIMC * RH * DH + 255) / 256, 256, 0, stream>>>(
      Wrv, WRVT, DIMC, RH * DH, 1.0f);
  transpose_scale_kernel<<<(DIMC * DIMC + 255) / 256, 256, 0, stream>>>(
      Wrq, WRQT, DIMC, DIMC, scale);
  transpose_scale_kernel<<<(DH * DH + 255) / 256, 256, 0, stream>>>(
      Wrk, WRKT, DH, DH, 1.0f);
  transpose_scale_kernel<<<(DIMC * DIMC + 255) / 256, 256, 0, stream>>>(
      Wout, WOUTT, DIMC, DIMC, 1.0f);

  // 2) projections (WMMA GEMMs): sq*scale, sk, rq*scale into (b,h,n,d); rv transposed
  gemm_bf16_wmma<<<(BN / 64) * (DIMC / 64), 128, 0, stream>>>(
      XBF, DIMC, WSQT, DIMC, BN, DIMC, DIMC, SQ, /*mode=*/1, SH);
  gemm_bf16_wmma<<<(BN / 64) * (DIMC / 64), 128, 0, stream>>>(
      XBF, DIMC, WSKT, DIMC, BN, DIMC, DIMC, SK, /*mode=*/1, SH);
  gemm_bf16_wmma<<<(BN / 64) * (DIMC / 64), 128, 0, stream>>>(
      XBF, DIMC, WRQT, DIMC, BN, DIMC, DIMC, RQ, /*mode=*/1, SH);
  gemm_bf16_wmma<<<(BN / 64) * ((RH * DH) / 64), 128, 0, stream>>>(
      XBF, DIMC, WRVT, DIMC, BN, RH * DH, DIMC, RVT, /*mode=*/2, RH);

  // 3) fused flash search attention -> retrieved (b,s,r,n,d) bf16
  flash_search_kernel<<<BATCH * SH * (NSEQ / 128), 256, 0, stream>>>(SQ, SK,
                                                                     RVT, RET);

  // 4) rk = retrieved @ Wrk  (M = b*s*r*n = 65536, K = N = 64)
  gemm_bf16_wmma<<<((BATCH * SH * RH * NSEQ) / 64) * (DH / 64), 128, 0,
                   stream>>>(RET, DH, WRKT, DH, BATCH * SH * RH * NSEQ, DH, DH,
                             RK, /*mode=*/0, 1);

  // 5) retrieval softmax over r + combine -> O (b,n,s*d) bf16
  combine_kernel<<<(BATCH * SH * NSEQ) / 8, 256, 0, stream>>>(RQ, RET, RK, O);

  // 6) out = O @ Wout -> f32 d_out (b,n,512)
  gemm_bf16_wmma<<<(BN / 64) * (DIMC / 64), 128, 0, stream>>>(
      O, DIMC, WOUTT, DIMC, BN, DIMC, DIMC, (float*)d_out, /*mode=*/3, 1);
}
